// EdgeClassifier_4020089389438
// MI455X (gfx1250) — matrix operand order repaired and checked by reading, not compiled, not run on hardware
//
#include <hip/hip_runtime.h>
#include <math.h>

// ---------------------------------------------------------------------------
// Types for WMMA fragments (gfx1250, wave32)
// ---------------------------------------------------------------------------
typedef __attribute__((ext_vector_type(16))) __bf16 v16bf;
typedef __attribute__((ext_vector_type(8)))  float  v8f;
typedef unsigned int u32x4 __attribute__((ext_vector_type(4)));

union Frag {
    v16bf          v;
    unsigned short u[16];
    u32x4          q[2];
};

__device__ __forceinline__ unsigned short f2bf(float f) {
    unsigned int u = __float_as_uint(f);
    u += 0x7FFFu + ((u >> 16) & 1u);      // round-to-nearest-even
    return (unsigned short)(u >> 16);
}
__device__ __forceinline__ float bf2f(unsigned short u) {
    return __uint_as_float(((unsigned int)u) << 16);
}

// A fragment (16x32 bf16), row-major source with stride 128.
// Lane l: M = l&15, half = l>>4.
//   elems 0..15 = A[M][kbase + half*16 + j]
__device__ __forceinline__ v16bf load_a_frag(const unsigned short* base,
                                             int kbase, int lane) {
    int m = lane & 15, half = lane >> 4;
    Frag f;
    const unsigned short* r = base + m * 128 + kbase + half * 16;
    f.q[0] = *(const u32x4*)(r);
    f.q[1] = *(const u32x4*)(r + 8);
    return f.v;
}

// B fragment from pre-swizzled weight buffer: contiguous 32B per lane.
// Layout: wb[(((ntile*4)+kt)*32 + lane)*16 + j]  where fragment elem j of
// lane l equals W[kt*32 + (l>>4)*16 + j][ntile*16 + (l&15)].
__device__ __forceinline__ v16bf load_b_frag(const unsigned short* wb,
                                             int ntile, int kt, int lane) {
    Frag f;
    const unsigned short* p = wb + (((ntile * 4) + kt) * 32 + lane) * 16;
    f.q[0] = *(const u32x4*)(p);
    f.q[1] = *(const u32x4*)(p + 8);
    return f.v;
}

#define WMMA_BF16(A, B, C) \
    __builtin_amdgcn_wmma_f32_16x16x32_bf16(false, (A), false, (B), (short)0, (C), false, false)

// ---------------------------------------------------------------------------
// Degree / normalization
// ---------------------------------------------------------------------------
__global__ void k_deg_init(float* dinv, int N) {
    int n = blockIdx.x * 256 + threadIdx.x;
    if (n < N) dinv[n] = 1.0f;                 // self-loop
}
__global__ void k_deg_count(const int* __restrict__ dstI, float* dinv, int E) {
    int e = blockIdx.x * 256 + threadIdx.x;
    if (e < E) unsafeAtomicAdd(&dinv[dstI[e]], 1.0f);
}
__global__ void k_deg_rsqrt(float* dinv, int N) {
    int n = blockIdx.x * 256 + threadIdx.x;
    if (n < N) dinv[n] = rsqrtf(dinv[n]);
}

// ---------------------------------------------------------------------------
// Layer-1 linear: out[N,128] = x[N,4] @ W1[4,128]
// ---------------------------------------------------------------------------
__global__ void k_lin1(const float* __restrict__ x, const float* __restrict__ W1,
                       float* __restrict__ out, int N) {
    int n = blockIdx.x * 2 + (threadIdx.x >> 7);
    int h = threadIdx.x & 127;
    if (n >= N) return;
    float acc = 0.f;
#pragma unroll
    for (int j = 0; j < 4; ++j) acc += x[n * 4 + j] * W1[j * 128 + h];
    out[(size_t)n * 128 + h] = acc;
}

// ---------------------------------------------------------------------------
// Edge scatter: agg[dst] += lin[src] * dinv[src]*dinv[dst]   (one wave/edge)
// Accumulator (N*128*4B = 51MB) stays resident in the 192MB L2.
// ---------------------------------------------------------------------------
__global__ void k_scatter(const float* __restrict__ lin, const int* __restrict__ srcI,
                          const int* __restrict__ dstI, const float* __restrict__ dinv,
                          float* __restrict__ agg, int E) {
    int e    = blockIdx.x * 8 + (threadIdx.x >> 5);
    int lane = threadIdx.x & 31;
    if (e >= E) return;
    int s = srcI[e], d = dstI[e];
    float w = dinv[s] * dinv[d];
    const float* ls = lin + (size_t)s * 128;
    float*       ad = agg + (size_t)d * 128;
#pragma unroll
    for (int t = 0; t < 4; ++t) {
        int h = lane + t * 32;
        unsafeAtomicAdd(&ad[h], ls[h] * w);
    }
}

// ---------------------------------------------------------------------------
// Finalize: h = [relu](agg + lin*dinv^2 + b); store bf16
// ---------------------------------------------------------------------------
__global__ void k_finalize(const float* __restrict__ agg, const float* __restrict__ lin,
                           const float* __restrict__ dinv, const float* __restrict__ bias,
                           unsigned short* __restrict__ hb, int doRelu, int N) {
    int n = blockIdx.x * 2 + (threadIdx.x >> 7);
    int h = threadIdx.x & 127;
    if (n >= N) return;
    float di = dinv[n];
    size_t i = (size_t)n * 128 + h;
    float v  = agg[i] + lin[i] * di * di + bias[h];
    if (doRelu) v = fmaxf(v, 0.f);
    hb[i] = f2bf(v);
}

// ---------------------------------------------------------------------------
// Weight conversion + swizzle into B-fragment layout (bf16).
// mat 0: W2 (rows 0..127)   mat 1: Wm1 rows 0..127 (src part)
// mat 2: Wm1 rows 128..255  mat 3: Wm2
// ---------------------------------------------------------------------------
__global__ void k_cvt_weights(const float* __restrict__ W2, const float* __restrict__ Wm1,
                              const float* __restrict__ Wm2,
                              unsigned short* w2b, unsigned short* wm1s,
                              unsigned short* wm1d, unsigned short* wm2b) {
    int gid = blockIdx.x * 256 + threadIdx.x;     // 4 * 16384 total
    int mat = gid >> 14;
    int idx = gid & 16383;
    int j     = idx & 15;
    int lane  = (idx >> 4) & 31;
    int kt    = (idx >> 9) & 3;
    int ntile = (idx >> 11) & 7;
    int k = kt * 32 + (lane >> 4) * 16 + j;
    int n = ntile * 16 + (lane & 15);
    const float* src; unsigned short* dst; int roff = 0;
    switch (mat) {
        case 0:  src = W2;  dst = w2b;  break;
        case 1:  src = Wm1; dst = wm1s; break;
        case 2:  src = Wm1; dst = wm1d; roff = 128; break;
        default: src = Wm2; dst = wm2b; break;
    }
    dst[idx] = f2bf(src[(size_t)(roff + k) * 128 + n]);
}

// ---------------------------------------------------------------------------
// GEMM: out[N,128] = h1b[N,128] @ W2b   (WMMA bf16, f32 accum)
// One block = 16 nodes; 8 waves, wave w owns output cols [16w, 16w+16).
// Two independent accumulator chains hide the WMMA->WMMA hazard NOPs.
// ---------------------------------------------------------------------------
__global__ void k_gemm_w2(const unsigned short* __restrict__ hb,
                          const unsigned short* __restrict__ w2b,
                          float* __restrict__ out) {
    __shared__ __align__(16) unsigned short As[16 * 128];
    int tid  = threadIdx.x;
    int row0 = blockIdx.x * 16;
    // 256 threads x 16B = 4KB cooperative stage of the 16x128 A tile
    ((u32x4*)As)[tid] = ((const u32x4*)(hb + (size_t)row0 * 128))[tid];
    __syncthreads();

    int wave = tid >> 5, lane = tid & 31;
    v16bf a0 = load_a_frag(As, 0,  lane);
    v16bf a1 = load_a_frag(As, 32, lane);
    v16bf a2 = load_a_frag(As, 64, lane);
    v16bf a3 = load_a_frag(As, 96, lane);
    v16bf b0 = load_b_frag(w2b, wave, 0, lane);
    v16bf b1 = load_b_frag(w2b, wave, 1, lane);
    v16bf b2 = load_b_frag(w2b, wave, 2, lane);
    v16bf b3 = load_b_frag(w2b, wave, 3, lane);
    v8f acc0 = {}, acc1 = {};
    acc0 = WMMA_BF16(a0, b0, acc0);
    acc1 = WMMA_BF16(a1, b1, acc1);
    acc0 = WMMA_BF16(a2, b2, acc0);
    acc1 = WMMA_BF16(a3, b3, acc1);

    int n     = wave * 16 + (lane & 15);
    int mbase = (lane >> 4) * 8;
#pragma unroll
    for (int r = 0; r < 8; ++r)
        out[(size_t)(row0 + mbase + r) * 128 + n] = acc0[r] + acc1[r];
}

// ---------------------------------------------------------------------------
// Edge MLP: per 16-edge tile.
//   z1 = relu(h2[src]@Wm1s + h2[dst]@Wm1d + ea@Wm1e + bm1)   (WMMA + VALU tail)
//   z2 = relu(z1@Wm2 + bm2)                                   (WMMA)
//   out = sigmoid(z2 . wm3 + bm3)
// Layer 1 uses 4 independent WMMA chains, layer 2 uses 2, so the scheduler
// can cover the 5-slot BF16 WMMA RAW hazard without v_nop padding.
// ---------------------------------------------------------------------------
__global__ void k_edge_mlp(const unsigned short* __restrict__ h2b,
                           const float* __restrict__ edge_attr,
                           const int* __restrict__ srcI, const int* __restrict__ dstI,
                           const float* __restrict__ Wm1,   // f32, for rows 256..260
                           const float* __restrict__ bm1,
                           const unsigned short* __restrict__ wm1s,
                           const unsigned short* __restrict__ wm1d,
                           const unsigned short* __restrict__ wm2b,
                           const float* __restrict__ bm2,
                           const float* __restrict__ wm3,
                           const float* __restrict__ bm3,
                           float* __restrict__ out, int E) {
    __shared__ __align__(16) unsigned short As[16 * 128];
    __shared__ __align__(16) unsigned short Ad[16 * 128];
    __shared__ __align__(16) unsigned short Z [16 * 128];
    __shared__ float EA[16 * 5];
    __shared__ float RED[256];

    int tid = threadIdx.x;
    int e0  = blockIdx.x * 16;

    // Gather endpoint embeddings: thread -> (row 0..15, 16B segment 0..15)
    {
        int row = tid >> 4, seg = tid & 15;
        int e = e0 + row;
        int s = srcI[e], d = dstI[e];
        *(u32x4*)(As + row * 128 + seg * 8) = *(const u32x4*)(h2b + (size_t)s * 128 + seg * 8);
        *(u32x4*)(Ad + row * 128 + seg * 8) = *(const u32x4*)(h2b + (size_t)d * 128 + seg * 8);
    }
    if (tid < 80) EA[tid] = edge_attr[(size_t)e0 * 5 + tid];
    __syncthreads();

    int wave  = tid >> 5, lane = tid & 31;
    int ncol  = wave * 16 + (lane & 15);
    int mbase = (lane >> 4) * 8;

    // ---- layer 1: 8 WMMAs across 4 independent accumulator chains ----
    v8f c0 = {}, c1 = {}, c2 = {}, c3 = {};
#pragma unroll
    for (int kk = 0; kk < 2; ++kk) {
        int kt0 = kk * 2, kt1 = kk * 2 + 1;
        v16bf as0 = load_a_frag(As, kt0 * 32, lane);
        v16bf as1 = load_a_frag(As, kt1 * 32, lane);
        v16bf ad0 = load_a_frag(Ad, kt0 * 32, lane);
        v16bf ad1 = load_a_frag(Ad, kt1 * 32, lane);
        v16bf bs0 = load_b_frag(wm1s, wave, kt0, lane);
        v16bf bs1 = load_b_frag(wm1s, wave, kt1, lane);
        v16bf bd0 = load_b_frag(wm1d, wave, kt0, lane);
        v16bf bd1 = load_b_frag(wm1d, wave, kt1, lane);
        c0 = WMMA_BF16(as0, bs0, c0);
        c1 = WMMA_BF16(ad0, bd0, c1);
        c2 = WMMA_BF16(as1, bs1, c2);
        c3 = WMMA_BF16(ad1, bd1, c3);
    }
    float bias1 = bm1[ncol];
    float we[5];
#pragma unroll
    for (int j = 0; j < 5; ++j) we[j] = Wm1[(size_t)(256 + j) * 128 + ncol];
#pragma unroll
    for (int r = 0; r < 8; ++r) {
        int m = mbase + r;
        float v = (c0[r] + c1[r]) + (c2[r] + c3[r]) + bias1;
#pragma unroll
        for (int j = 0; j < 5; ++j) v += EA[m * 5 + j] * we[j];
        Z[m * 128 + ncol] = f2bf(fmaxf(v, 0.f));
    }
    __syncthreads();

    // ---- layer 2: 4 WMMAs across 2 independent chains ----
    v16bf z0 = load_a_frag(Z, 0,  lane);
    v16bf z1 = load_a_frag(Z, 32, lane);
    v16bf z2 = load_a_frag(Z, 64, lane);
    v16bf z3 = load_a_frag(Z, 96, lane);
    v16bf w0 = load_b_frag(wm2b, wave, 0, lane);
    v16bf w1 = load_b_frag(wm2b, wave, 1, lane);
    v16bf w2 = load_b_frag(wm2b, wave, 2, lane);
    v16bf w3 = load_b_frag(wm2b, wave, 3, lane);
    v8f d0 = {}, d1 = {};
    d0 = WMMA_BF16(z0, w0, d0);
    d1 = WMMA_BF16(z1, w1, d1);
    d0 = WMMA_BF16(z2, w2, d0);
    d1 = WMMA_BF16(z3, w3, d1);

    float bias2 = bm2[ncol];
#pragma unroll
    for (int r = 0; r < 8; ++r)                      // reuse As as z2 buffer
        As[(mbase + r) * 128 + ncol] = f2bf(fmaxf(d0[r] + d1[r] + bias2, 0.f));
    __syncthreads();

    // ---- layer 3: 128 -> 1 dot, sigmoid ----
    {
        int e = tid >> 4, c = tid & 15;              // 16 partials per edge
        float p = 0.f;
#pragma unroll
        for (int j = 0; j < 8; ++j) {
            int k = c * 8 + j;
            p += bf2f(As[e * 128 + k]) * wm3[k];
        }
        RED[tid] = p;
    }
    __syncthreads();
    if (tid < 16) {
        float s = bm3[0];
#pragma unroll
        for (int j = 0; j < 16; ++j) s += RED[tid * 16 + j];
        if (e0 + tid < E) out[e0 + tid] = 1.f / (1.f + __expf(-s));
    }
}

// ---------------------------------------------------------------------------
// Host launcher
// ---------------------------------------------------------------------------
static inline size_t alignup(size_t x) { return (x + 255) & ~(size_t)255; }

extern "C" void kernel_launch(void* const* d_in, const int* in_sizes, int n_in,
                              void* d_out, int out_size, void* d_ws, size_t ws_size,
                              hipStream_t stream) {
    const float* x   = (const float*)d_in[0];
    const float* ea  = (const float*)d_in[1];
    const float* W1  = (const float*)d_in[2];
    const float* b1  = (const float*)d_in[3];
    const float* W2  = (const float*)d_in[4];
    const float* b2  = (const float*)d_in[5];
    const float* Wm1 = (const float*)d_in[6];
    const float* bm1 = (const float*)d_in[7];
    const float* Wm2 = (const float*)d_in[8];
    const float* bm2 = (const float*)d_in[9];
    const float* wm3 = (const float*)d_in[10];
    const float* bm3 = (const float*)d_in[11];
    const int*  eidx = (const int*)d_in[12];

    int N = in_sizes[0] / 4;
    int E = in_sizes[1] / 5;
    const int* srcI = eidx;
    const int* dstI = eidx + E;

    // Workspace carve
    char* p = (char*)d_ws;
    float* bufA = (float*)p;                 p += alignup((size_t)N * 128 * 4);  // lin (layer1, then layer2)
    float* bufB = (float*)p;                 p += alignup((size_t)N * 128 * 4);  // agg accumulator
    unsigned short* h1b = (unsigned short*)p; p += alignup((size_t)N * 128 * 2);
    unsigned short* h2b = (unsigned short*)p; p += alignup((size_t)N * 128 * 2);
    float* dinv = (float*)p;                 p += alignup((size_t)N * 4);
    unsigned short* w2b  = (unsigned short*)p; p += alignup(16384 * 2);
    unsigned short* wm1s = (unsigned short*)p; p += alignup(16384 * 2);
    unsigned short* wm1d = (unsigned short*)p; p += alignup(16384 * 2);
    unsigned short* wm2b = (unsigned short*)p; p += alignup(16384 * 2);
    (void)ws_size; (void)n_in; (void)out_size;

    // 1) symmetric normalization coefficients
    k_deg_init <<<(N + 255) / 256, 256, 0, stream>>>(dinv, N);
    k_deg_count<<<(E + 255) / 256, 256, 0, stream>>>(dstI, dinv, E);
    k_deg_rsqrt<<<(N + 255) / 256, 256, 0, stream>>>(dinv, N);

    // 2) weight bf16 conversion + fragment swizzle
    k_cvt_weights<<<256, 256, 0, stream>>>(W2, Wm1, Wm2, w2b, wm1s, wm1d, wm2b);

    // 3) GCN layer 1
    k_lin1<<<(N + 1) / 2, 256, 0, stream>>>(x, W1, bufA, N);
    hipMemsetAsync(bufB, 0, (size_t)N * 128 * 4, stream);
    k_scatter<<<(E + 7) / 8, 256, 0, stream>>>(bufA, srcI, dstI, dinv, bufB, E);
    k_finalize<<<(N + 1) / 2, 256, 0, stream>>>(bufB, bufA, dinv, b1, h1b, 1, N);

    // 4) GCN layer 2  (N assumed multiple of 16; N = 100000 = 6250*16)
    k_gemm_w2<<<N / 16, 256, 0, stream>>>(h1b, w2b, bufA);
    hipMemsetAsync(bufB, 0, (size_t)N * 128 * 4, stream);
    k_scatter<<<(E + 7) / 8, 256, 0, stream>>>(bufA, srcI, dstI, dinv, bufB, E);
    k_finalize<<<(N + 1) / 2, 256, 0, stream>>>(bufB, bufA, dinv, b2, h2b, 0, N);

    // 5) edge MLP  (E assumed multiple of 16; E = 600000 = 37500*16)
    k_edge_mlp<<<E / 16, 256, 0, stream>>>(h2b, ea, srcI, dstI, Wm1, bm1,
                                           wm1s, wm1d, wm2b, bm2, wm3, bm3,
                                           (float*)d_out, E);
}
